// TritonModel_2078764171682
// MI455X (gfx1250) — matrix operand order repaired and checked by reading, not compiled, not run on hardware
//
#include <hip/hip_runtime.h>
#include <hip/hip_bf16.h>

#define HIDDEN 5120
#define HD 256
#define ROPE_DIM 64
#define QROWS 6144       // 24*256
#define KVROWS 1024      // 4*256

typedef __attribute__((ext_vector_type(8)))  __bf16 v8bf;
typedef __attribute__((ext_vector_type(16))) __bf16 v16bf;
typedef __attribute__((ext_vector_type(8)))  float  v8f;

__global__ __launch_bounds__(128)
void qkv_rope_wmma(const __hip_bfloat16* __restrict__ x,
                   const __hip_bfloat16* __restrict__ cosp,
                   const __hip_bfloat16* __restrict__ sinp,
                   const __hip_bfloat16* __restrict__ wq,
                   const __hip_bfloat16* __restrict__ wk,
                   const __hip_bfloat16* __restrict__ wv,
                   __hip_bfloat16* __restrict__ out)
{
    const int lane = threadIdx.x & 31;
    const int wave = threadIdx.x >> 5;
    const int tile = blockIdx.x * 4 + wave;   // 0..511
    const int n0   = tile * 16;               // base output row of this tile

    // Select weight matrix (tile never straddles a matrix: 6144,1024 are multiples of 16)
    const __hip_bfloat16* W;
    int row0;
    if (n0 < QROWS)              { W = wq; row0 = n0; }
    else if (n0 < QROWS+KVROWS)  { W = wk; row0 = n0 - QROWS; }
    else                         { W = wv; row0 = n0 - QROWS - KVROWS; }

    const int mrow = lane & 15;   // M row within tile (A layout)
    const int half = lane >> 4;   // 0: K {0..7,16..23} / 1: K {8..15,24..31}

    const __bf16* rowp = (const __bf16*)W + (size_t)(row0 + mrow) * HIDDEN;
    const __bf16* xp   = (const __bf16*)x;

    const int aoff = half * 8;    // A k-offset within 32-chunk (elements)
    const int boff = half * 8;    // B k-offset within 32-chunk (elements)

    v8f acc = {};

    #pragma unroll 4
    for (int k0 = 0; k0 < HIDDEN; k0 += 32) {
        // A fragment: 16x32 bf16 weight tile, 2x16B contiguous loads per lane
        v8bf alo = *(const v8bf*)(rowp + k0 + aoff);
        v8bf ahi = *(const v8bf*)(rowp + k0 + aoff + 16);
        // B fragment: x chunk broadcast into all 16 columns (every column of D
        // then holds the same GEMV partial; we consume column 0). 32B per lane,
        // identical address within each 16-lane half -> L0 broadcast.
        v8bf xlo = *(const v8bf*)(xp + k0 + boff);
        v8bf xhi = *(const v8bf*)(xp + k0 + boff + 16);

        v16bf A = __builtin_shufflevector(alo, ahi,
                     0,1,2,3,4,5,6,7,8,9,10,11,12,13,14,15);
        v16bf B = __builtin_shufflevector(xlo, xhi,
                     0,1,2,3,4,5,6,7,8,9,10,11,12,13,14,15);

        acc = __builtin_amdgcn_wmma_f32_16x16x32_bf16(
                  false, A, false, B, (short)0, acc, false, false);
    }

    // D column 0: lane 0 holds rows n0+0..7 in acc[0..7], lane 16 holds n0+8..15.
    if (mrow == 0) {
        const int orow = n0 + half * 8;          // flat output index (q|k|v concat)
        const int d0   = orow & (HD - 1);        // dim within head
        const bool do_rope = (n0 < QROWS + KVROWS) && (d0 < ROPE_DIM);

        alignas(16) __hip_bfloat16 o[8];
        if (do_rope) {
            const int jb = d0 >> 1;              // pair index base
            #pragma unroll
            for (int p = 0; p < 4; ++p) {
                // match reference: fp32 accum -> bf16 -> fp32 -> rotate -> bf16
                float e  = __bfloat162float(__float2bfloat16(acc[2*p]));
                float od = __bfloat162float(__float2bfloat16(acc[2*p+1]));
                float cf = __bfloat162float(cosp[jb + p]);
                float sf = __bfloat162float(sinp[jb + p]);
                o[2*p]   = __float2bfloat16(e * cf - od * sf);
                o[2*p+1] = __float2bfloat16(e * sf + od * cf);
            }
        } else {
            #pragma unroll
            for (int i = 0; i < 8; ++i) o[i] = __float2bfloat16(acc[i]);
        }
        *(uint4*)((__hip_bfloat16*)out + orow) = *(const uint4*)o;
    }
}

extern "C" void kernel_launch(void* const* d_in, const int* in_sizes, int n_in,
                              void* d_out, int out_size, void* d_ws, size_t ws_size,
                              hipStream_t stream) {
    const __hip_bfloat16* x    = (const __hip_bfloat16*)d_in[0];
    const __hip_bfloat16* cosp = (const __hip_bfloat16*)d_in[1];
    const __hip_bfloat16* sinp = (const __hip_bfloat16*)d_in[2];
    const __hip_bfloat16* wq   = (const __hip_bfloat16*)d_in[3];
    const __hip_bfloat16* wk   = (const __hip_bfloat16*)d_in[4];
    const __hip_bfloat16* wv   = (const __hip_bfloat16*)d_in[5];
    __hip_bfloat16* out = (__hip_bfloat16*)d_out;

    // 512 wave-tiles: 128 blocks x 4 waves (128 threads, wave32)
    qkv_rope_wmma<<<dim3(128), dim3(128), 0, stream>>>(x, cosp, sinp, wq, wk, wv, out);
}
